// MultiFieldEncoder_3633542333065
// MI455X (gfx1250) — compile-verified, matrix operand order checked by reading, not run on hardware
//
#include <hip/hip_runtime.h>
#include <math.h>

typedef _Float16 v8h  __attribute__((ext_vector_type(8)));
typedef _Float16 v16h __attribute__((ext_vector_type(16)));
typedef float    v8f  __attribute__((ext_vector_type(8)));

#define NB 4
#define NN 40000
#define NEDGE 160000
#define ND 128
#define VOFF (2L*4*40000*128)   // offset of E_enc in d_out (floats)
#define WPK_ELEMS 16384         // halves per packed 128x128 matrix (32 KB)

__device__ __forceinline__ float silu_f(float x){ return x / (1.0f + __expf(-x)); }

// sum across a 16-lane half-wave (masks <16 keep the two halves independent)
__device__ __forceinline__ float hsum16(float v){
  v += __shfl_xor(v, 1, 32);
  v += __shfl_xor(v, 2, 32);
  v += __shfl_xor(v, 4, 32);
  v += __shfl_xor(v, 8, 32);
  return v;
}

// 32-bit LDS byte offset of a shared-memory pointer (AS3 pointers are LDS offsets)
__device__ __forceinline__ uint32_t lds_off(const void* p){
  return (uint32_t)(uintptr_t)(const __attribute__((address_space(3))) char*)p;
}

// gfx1250 async copy: 16B global -> LDS, tracked with ASYNCcnt (ISA 08 §4, GV mode)
__device__ __forceinline__ void async_load_b128(uint32_t lds_addr, const void* gaddr){
  asm volatile("global_load_async_to_lds_b128 %0, %1, off"
               :: "v"(lds_addr), "v"(gaddr) : "memory");
}
__device__ __forceinline__ void wait_asynccnt0(){
  asm volatile("s_wait_asynccnt 0x0" ::: "memory");
}

// ---------------------------------------------------------------------------
// Prepack (run once per launch): 128x128 row-major f32 W[k][n] -> wave32 WMMA
// B-fragment f16 layout dst[ks][nt][lane][e]:
//   lane&15 -> N column within 16-wide N-tile, element e:
//   K = 32*ks + ((e<8 ? e : e+8) + 8*(lane>>4))
// dst order: [0]=fW2 f0, [1]=fW2 f1, [2]=xW f0, [3]=xW f1, [4]=eW2
// ---------------------------------------------------------------------------
__global__ void __launch_bounds__(256)
prepack_kernel(const float* __restrict__ fW2, const float* __restrict__ xW,
               const float* __restrict__ eW2, _Float16* __restrict__ dst)
{
  const int idx = blockIdx.x*256 + threadIdx.x;
  if (idx >= 5*WPK_ELEMS) return;
  const int m = idx >> 14;
  const int r = idx & (WPK_ELEMS-1);
  const float* src = (m==0) ? fW2 : (m==1) ? (fW2 + WPK_ELEMS)
                   : (m==2) ? xW  : (m==3) ? (xW  + WPK_ELEMS) : eW2;
  const int e    = r & 15;
  const int lane = (r >> 4) & 31;
  const int nt   = (r >> 9) & 7;
  const int ks   = (r >> 12) & 3;
  const int k = 32*ks + ((e < 8 ? e : e + 8) + (lane >> 4)*8);
  const int n = 16*nt + (lane & 15);
  dst[idx] = (_Float16)src[k*ND + n];
}

// One wave computes a 16x128 output tile:  acc[nt] += Arow(16x128) @ W(128x128)
// arow: f16 row-major [16][128] slice of LDS (this wave's 16 rows, row = lane&15)
// wbase: packed weights for this matrix (16384 halves, already in LDS)
__device__ __forceinline__ void gemm_16x128(const _Float16* __restrict__ arow,
                                            const _Float16* __restrict__ wbase,
                                            int lane, v8f acc[8]){
  const int half8 = (lane >> 4) * 8;
  #pragma unroll
  for (int ks = 0; ks < 4; ks++){
    const _Float16* ap = arow + 32*ks + half8;     // two contiguous 8-half runs
    v8h alo = *(const v8h*)(ap);
    v8h ahi = *(const v8h*)(ap + 16);
    v16h a = __builtin_shufflevector(alo, ahi,
              0,1,2,3,4,5,6,7,8,9,10,11,12,13,14,15);
    const _Float16* wp = wbase + ks*4096 + lane*16;
    #pragma unroll
    for (int nt = 0; nt < 8; nt++){
      const _Float16* q = wp + nt*512;
      v8h blo = *(const v8h*)(q);
      v8h bhi = *(const v8h*)(q + 8);
      v16h bm = __builtin_shufflevector(blo, bhi,
                 0,1,2,3,4,5,6,7,8,9,10,11,12,13,14,15);
      acc[nt] = __builtin_amdgcn_wmma_f32_16x16x32_f16(
          /*neg_a=*/false, a, /*neg_b=*/false, bm,
          /*c_mod=*/(short)0, acc[nt], /*reuse_a=*/false, /*reuse_b=*/false);
    }
  }
}

// ---------------------------------------------------------------------------
// FiLM parameter kernel: gamma/beta[b][128] -> gb[b*256 + {d, 128+d}]
// ---------------------------------------------------------------------------
__global__ void __launch_bounds__(256)
film_kernel(const float* __restrict__ time_i, const float* __restrict__ conditions,
            const float* __restrict__ tW1, const float* __restrict__ tb1,
            const float* __restrict__ tW2, const float* __restrict__ tb2,
            const float* __restrict__ cW1, const float* __restrict__ cb1,
            const float* __restrict__ cW2, const float* __restrict__ cb2,
            const float* __restrict__ pW1, const float* __restrict__ pb1,
            const float* __restrict__ pW2, const float* __restrict__ pb2,
            float* __restrict__ gb)
{
  __shared__ float t1[4][128], c1[4][128], te[4][128], ce[4][128], p1[4][128];
  const int t = threadIdx.x;
  if (t < 128){
    for (int b = 0; b < 4; b++){
      float s = tb1[t];
      for (int i = 0; i < 11; i++) s += time_i[b*11 + i]*tW1[i*128 + t];
      t1[b][t] = silu_f(s);
      float c = cb1[t];
      for (int i = 0; i < 12; i++) c += conditions[b*12 + i]*cW1[i*128 + t];
      c1[b][t] = silu_f(c);
    }
  }
  __syncthreads();
  if (t < 128){
    for (int b = 0; b < 4; b++){
      float s = tb2[t], c = cb2[t];
      for (int i = 0; i < 128; i++){
        s += t1[b][i]*tW2[i*128 + t];
        c += c1[b][i]*cW2[i*128 + t];
      }
      te[b][t] = s; ce[b][t] = c;
    }
  }
  __syncthreads();
  if (t < 128){
    for (int b = 0; b < 4; b++){
      float s = pb1[t];
      for (int i = 0; i < 128; i++){
        s += ce[b][i]*pW1[i*128 + t];            // concat order: [cond, time]
        s += te[b][i]*pW1[(128 + i)*128 + t];
      }
      p1[b][t] = silu_f(s);
    }
  }
  __syncthreads();
  for (int b = 0; b < 4; b++){
    float s = pb2[t];
    for (int i = 0; i < 128; i++) s += p1[b][i]*pW2[i*256 + t];
    gb[b*256 + t] = s;      // t<128 -> gamma, t>=128 -> beta
  }
}

// ---------------------------------------------------------------------------
// Node kernel: 64 nodes/block, both fields fused. Weights arrive via async
// global->LDS b128 copies of the prepacked image (overlapped with phase A).
// Dynamic LDS: packedW 64KB (fW2 both fields, then xW) + hbuf 32KB = 96KB.
// ---------------------------------------------------------------------------
__global__ void __launch_bounds__(256)
node_kernel(const float* __restrict__ node_pos, const float* __restrict__ state_in,
            const float* __restrict__ fW1, const float* __restrict__ fb1,
            const float* __restrict__ fb2,
            const float* __restrict__ fn_g, const float* __restrict__ fn_b,
            const float* __restrict__ xb,  const float* __restrict__ gate,
            const float* __restrict__ en_g, const float* __restrict__ en_b,
            const float* __restrict__ gb, const _Float16* __restrict__ wpk,
            float* __restrict__ out)
{
  extern __shared__ char smem[];
  _Float16* packedW = (_Float16*)smem;             // 2 * 16384 halves = 64 KB
  _Float16* hbuf    = (_Float16*)(smem + 65536);   // [2][64][128] f16 = 32 KB

  const int tid = threadIdx.x;
  const int rowBase = blockIdx.x * 64;             // 64 | 40000, batch-aligned
  const int b = rowBase / NN;
  const int nBase = rowBase - b * NN;
  const uint32_t ldsW = lds_off(packedW);

  // kick off async fetch of packed fW2 (both fields, 64 KB = 4096 x 16B)
  {
    const char* gsrc = (const char*)wpk;
    #pragma unroll
    for (int i = 0; i < 16; i++){
      const int chunk = tid + i*256;
      async_load_b128(ldsW + chunk*16, gsrc + chunk*16);
    }
  }

  // phase A (overlaps async): h = silu([x, pos] @ fW1 + fb1) -> f16 LDS
  for (int idx = tid; idx < 2*64*128; idx += 256){
    const int d  = idx & 127;
    const int rl = (idx >> 7) & 63;
    const int f  = idx >> 13;
    const int node = b * NN + nBase + rl;
    const float x  = state_in[node*2 + f];
    const float p0 = node_pos[node*3 + 0];
    const float p1 = node_pos[node*3 + 1];
    const float p2 = node_pos[node*3 + 2];
    const float* w = fW1 + f*512;
    float s = x*w[d] + p0*w[128+d] + p1*w[256+d] + p2*w[384+d] + fb1[f*128 + d];
    hbuf[(f*64 + rl)*128 + d] = (_Float16)silu_f(s);
  }
  wait_asynccnt0();
  __syncthreads();

  const int wave = tid >> 5;
  const int lane = tid & 31;
  const int f    = wave >> 2;          // waves 0-3: field 0, waves 4-7: field 1
  const int wrow = (wave & 3) * 16;    // 16-row tile within the 64-node block
  const int hl   = lane >> 4;
  const int dcol = lane & 15;          // N column within N-tile (and A row M)

  v8f acc[8] = {};
  gemm_16x128(hbuf + (f*64 + wrow + dcol)*128, packedW + f*WPK_ELEMS, lane, acc);

  // per-column parameters for FiLM + first LN
  float gg[8], bb[8], b2[8], ng[8], nb[8];
  #pragma unroll
  for (int nt = 0; nt < 8; nt++){
    const int d = 16*nt + dcol;
    gg[nt] = gb[b*256 + d];
    bb[nt] = gb[b*256 + 128 + d];
    b2[nt] = fb2[f*128 + d];
    ng[nt] = fn_g[f*128 + d];
    nb[nt] = fn_b[f*128 + d];
  }
  #pragma unroll
  for (int nt = 0; nt < 8; nt++)
    #pragma unroll
    for (int r = 0; r < 8; r++)
      acc[nt][r] = gg[nt]*(acc[nt][r] + b2[nt]) + bb[nt];

  // LayerNorm(fn): row M = r + 8*hl lives in one 16-lane half across 8 N-tiles
  #pragma unroll
  for (int r = 0; r < 8; r++){
    float s = 0.f;
    #pragma unroll
    for (int nt = 0; nt < 8; nt++) s += acc[nt][r];
    const float mu = hsum16(s) * (1.0f/128.0f);
    float v = 0.f;
    #pragma unroll
    for (int nt = 0; nt < 8; nt++){ float t = acc[nt][r] - mu; v += t*t; }
    const float rs = rsqrtf(hsum16(v)*(1.0f/128.0f) + 1e-5f);
    #pragma unroll
    for (int nt = 0; nt < 8; nt++)
      acc[nt][r] = (acc[nt][r] - mu)*rs*ng[nt] + nb[nt];
  }

  // publish V (f16) for the cross-field exchange (each wave owns its 16 rows)
  #pragma unroll
  for (int nt = 0; nt < 8; nt++)
    #pragma unroll
    for (int r = 0; r < 8; r++)
      hbuf[(f*64 + wrow + r + 8*hl)*128 + 16*nt + dcol] = (_Float16)acc[nt][r];
  __syncthreads();   // all GEMM1 reads of packedW and V stores complete

  // async swap of the weight buffer to packed xW (both fields)
  {
    const char* gsrc = (const char*)(wpk + 2*WPK_ELEMS);
    #pragma unroll
    for (int i = 0; i < 16; i++){
      const int chunk = tid + i*256;
      async_load_b128(ldsW + chunk*16, gsrc + chunk*16);
    }
  }
  wait_asynccnt0();
  __syncthreads();

  // exchange GEMM: other = V of the opposite field (F==2)
  v8f acc2[8] = {};
  gemm_16x128(hbuf + ((1 - f)*64 + wrow + dcol)*128, packedW + f*WPK_ELEMS,
              lane, acc2);

  float xbv[8], eg[8], ebv[8];
  #pragma unroll
  for (int nt = 0; nt < 8; nt++){
    const int d = 16*nt + dcol;
    xbv[nt] = xb[f*128 + d];
    eg[nt]  = en_g[f*128 + d];
    ebv[nt] = en_b[f*128 + d];
  }
  const float tg = tanhf(gate[f]);
  #pragma unroll
  for (int nt = 0; nt < 8; nt++)
    #pragma unroll
    for (int r = 0; r < 8; r++)
      acc2[nt][r] = acc[nt][r] + tg * silu_f(acc2[nt][r] + xbv[nt]);

  // LayerNorm(en) + coalesced f32 store of V_new [F,B,N,D]
  #pragma unroll
  for (int r = 0; r < 8; r++){
    float s = 0.f;
    #pragma unroll
    for (int nt = 0; nt < 8; nt++) s += acc2[nt][r];
    const float mu = hsum16(s) * (1.0f/128.0f);
    float v = 0.f;
    #pragma unroll
    for (int nt = 0; nt < 8; nt++){ float t = acc2[nt][r] - mu; v += t*t; }
    const float rs = rsqrtf(hsum16(v)*(1.0f/128.0f) + 1e-5f);
    const long n = nBase + wrow + r + 8*hl;
    float* op = out + (((long)(f*NB + b))*NN + n)*ND;
    #pragma unroll
    for (int nt = 0; nt < 8; nt++)
      op[16*nt + dcol] = (acc2[nt][r] - mu)*rs*eg[nt] + ebv[nt];
  }
}

// ---------------------------------------------------------------------------
// Edge kernel: 128 edges/block; geometry + 7->128 MLP on VALU, 128x128 on WMMA.
// eW2 arrives via async global->LDS b128 (32 KB), overlapped with phase A.
// Dynamic LDS: packedW 32KB + hbuf [128][128] f16 32KB = 64KB.
// ---------------------------------------------------------------------------
__global__ void __launch_bounds__(256)
edge_kernel(const float* __restrict__ node_pos, const int* __restrict__ edges,
            const float* __restrict__ eW1, const float* __restrict__ eb1,
            const float* __restrict__ eb2, const _Float16* __restrict__ wpk_e,
            float* __restrict__ out)
{
  extern __shared__ char smem[];
  _Float16* packedW = (_Float16*)smem;             // 16384 halves = 32 KB
  _Float16* hbuf    = (_Float16*)(smem + 32768);   // [128][128] f16 = 32 KB

  const int tid = threadIdx.x;
  {
    const uint32_t ldsW = lds_off(packedW);
    const char* gsrc = (const char*)wpk_e;         // 32 KB = 2048 x 16B
    #pragma unroll
    for (int i = 0; i < 8; i++){
      const int chunk = tid + i*256;
      async_load_b128(ldsW + chunk*16, gsrc + chunk*16);
    }
  }

  const int edgeBase = blockIdx.x * 128;           // 128 | 160000, batch-aligned
  const int b = edgeBase / NEDGE;
  {
    const int rl  = tid >> 1;                      // edge within tile
    const int idx = edgeBase + rl;
    const int si = edges[idx*2 + 0];
    const int ri = edges[idx*2 + 1];
    const float* ps = node_pos + ((long)b*NN + si)*3;
    const float* pr = node_pos + ((long)b*NN + ri)*3;
    const float d0 = pr[0]-ps[0], d1 = pr[1]-ps[1], d2 = pr[2]-ps[2];
    const float nrm = sqrtf(d0*d0 + d1*d1 + d2*d2);
    const float feat[7] = {d0, d1, d2, -d0, -d1, -d2, nrm};
    const int dbase = (tid & 1) * 64;
    for (int dd = 0; dd < 64; dd++){
      const int d = dbase + dd;
      float s = eb1[d];
      #pragma unroll
      for (int i = 0; i < 7; i++) s += feat[i]*eW1[i*ND + d];
      hbuf[rl*ND + d] = (_Float16)silu_f(s);
    }
  }
  wait_asynccnt0();
  __syncthreads();

  const int wave = tid >> 5;
  const int lane = tid & 31;
  const int hl   = lane >> 4;
  const int dcol = lane & 15;
  const int wrow = wave * 16;

  v8f acc[8] = {};
  gemm_16x128(hbuf + (wrow + dcol)*ND, packedW, lane, acc);

  #pragma unroll
  for (int nt = 0; nt < 8; nt++){
    const int d = 16*nt + dcol;
    const float bias = eb2[d];
    #pragma unroll
    for (int r = 0; r < 8; r++){
      const long idx = edgeBase + wrow + r + 8*hl;   // flat (b,e) row
      out[VOFF + idx*ND + d] = acc[nt][r] + bias;
    }
  }
}

// ---------------------------------------------------------------------------
extern "C" void kernel_launch(void* const* d_in, const int* in_sizes, int n_in,
                              void* d_out, int out_size, void* d_ws, size_t ws_size,
                              hipStream_t stream){
  (void)in_sizes; (void)n_in; (void)out_size; (void)ws_size;
  const float* node_pos   = (const float*)d_in[0];
  const float* state_in   = (const float*)d_in[1];
  const float* time_i     = (const float*)d_in[2];
  const float* conditions = (const float*)d_in[3];
  const int*   edges      = (const int*)d_in[4];
  const float* fW1 = (const float*)d_in[5];
  const float* fb1 = (const float*)d_in[6];
  const float* fW2 = (const float*)d_in[7];
  const float* fb2 = (const float*)d_in[8];
  const float* tW1 = (const float*)d_in[9];
  const float* tb1 = (const float*)d_in[10];
  const float* tW2 = (const float*)d_in[11];
  const float* tb2 = (const float*)d_in[12];
  const float* cW1 = (const float*)d_in[13];
  const float* cb1 = (const float*)d_in[14];
  const float* cW2 = (const float*)d_in[15];
  const float* cb2 = (const float*)d_in[16];
  const float* pW1 = (const float*)d_in[17];
  const float* pb1 = (const float*)d_in[18];
  const float* pW2 = (const float*)d_in[19];
  const float* pb2 = (const float*)d_in[20];
  const float* eW1 = (const float*)d_in[21];
  const float* eb1 = (const float*)d_in[22];
  const float* eW2 = (const float*)d_in[23];
  const float* eb2 = (const float*)d_in[24];
  const float* fn_g = (const float*)d_in[25];
  const float* fn_b = (const float*)d_in[26];
  const float* en_g = (const float*)d_in[27];
  const float* en_b = (const float*)d_in[28];
  const float* xW   = (const float*)d_in[29];
  const float* xb   = (const float*)d_in[30];
  const float* gate = (const float*)d_in[31];

  float*     gb     = (float*)d_ws;                       // gamma/beta [B][256]
  _Float16*  packed = (_Float16*)((char*)d_ws + 4096);    // 5 * 32 KB packed W
  float*     out    = (float*)d_out;

  film_kernel<<<1, 256, 0, stream>>>(time_i, conditions, tW1,tb1,tW2,tb2,
                                     cW1,cb1,cW2,cb2, pW1,pb1,pW2,pb2, gb);
  prepack_kernel<<<(5*WPK_ELEMS + 255)/256, 256, 0, stream>>>(fW2, xW, eW2, packed);
  node_kernel<<<(NB*NN)/64, 256, 96*1024, stream>>>(
      node_pos, state_in, fW1,fb1, fb2, fn_g,fn_b, xb, gate, en_g,en_b,
      gb, packed, out);
  edge_kernel<<<(NB*NEDGE)/128, 256, 64*1024, stream>>>(
      node_pos, edges, eW1,eb1, eb2, packed + 4*WPK_ELEMS, out);
}
